// AverageAttention_58626303590878
// MI455X (gfx1250) — matrix operand order repaired and checked by reading, not compiled, not run on hardware
//
#include <hip/hip_runtime.h>
#include <hip/hip_bf16.h>
#include <cstdint>

typedef __attribute__((ext_vector_type(16))) __bf16 v16bf;
typedef __attribute__((ext_vector_type(8)))  float  v8f;
typedef __attribute__((ext_vector_type(4)))  int    v4i_vec;

#define NTOK   16384   // 8 * 2048
#define DM     1024
#define SEQ    2048
#define NBATCH 8

#define AS1 __attribute__((address_space(1)))
#define AS3 __attribute__((address_space(3)))

union AFrag { v16bf v; unsigned int u[8]; uint4 q[2]; };

// ---------------------------------------------------------- async copy path
#if defined(__has_builtin)
#if __has_builtin(__builtin_amdgcn_global_load_async_to_lds_b128)
#define HAVE_ASYNC_BUILTIN 1
#endif
#if __has_builtin(__builtin_amdgcn_s_wait_asynccnt)
#define HAVE_ASYNC_WAIT_BUILTIN 1
#endif
#endif

__device__ __forceinline__ void async_cp16(const __bf16* g, __bf16* l) {
#ifdef HAVE_ASYNC_BUILTIN
  __builtin_amdgcn_global_load_async_to_lds_b128(
      (AS1 v4i_vec*)g, (AS3 v4i_vec*)l, 0, 0);
#else
  unsigned laddr = (unsigned)(unsigned long long)l;  // LDS byte offset
  asm volatile("global_load_async_to_lds_b128 %0, %1, off"
               :: "v"(laddr), "v"(g) : "memory");
#endif
}

__device__ __forceinline__ void async_wait0() {
#ifdef HAVE_ASYNC_WAIT_BUILTIN
  __builtin_amdgcn_s_wait_asynccnt(0);
#else
  asm volatile("s_wait_asynccnt 0x0" ::: "memory");
#endif
}

// ------------------------------------- one-time weight convert + transpose
// src: f32 [K][N] row-major -> dst: bf16 [N][K] row-major
__global__ __launch_bounds__(256) void transpose_bf16_kernel(const float* __restrict__ src,
                                                             __bf16* __restrict__ dst,
                                                             int K, int N) {
  __shared__ float t[32][33];
  int k0 = blockIdx.x * 32, n0 = blockIdx.y * 32;
  int tx = threadIdx.x & 31, ty = threadIdx.x >> 5;
#pragma unroll
  for (int p = 0; p < 4; ++p)
    t[ty + 8 * p][tx] = src[(size_t)(k0 + ty + 8 * p) * N + n0 + tx];
  __syncthreads();
#pragma unroll
  for (int p = 0; p < 4; ++p)
    dst[(size_t)(n0 + ty + 8 * p) * K + k0 + tx] = (__bf16)t[tx][ty + 8 * p];
}

// ------------------------------------------------- stage 1: chunked cumsum
// grid = 8 batches * 16 seq-chunks(128) * 4 d-groups(256) = 512 blocks
__global__ __launch_bounds__(256) void cumsum_chunk_kernel(const float* __restrict__ x,
                                                           float* __restrict__ cum,
                                                           __bf16* __restrict__ xbf,
                                                           float* __restrict__ chunksum) {
  int blk  = blockIdx.x;
  int dgrp = blk & 3;
  int chnk = (blk >> 2) & 15;
  int b    = blk >> 6;
  int d    = dgrp * 256 + threadIdx.x;
  size_t base = ((size_t)b * SEQ + (size_t)chnk * 128) * DM + d;
  float acc = 0.f;
  for (int s = 0; s < 128; ++s) {
    size_t idx = base + (size_t)s * DM;
    float v = x[idx];
    acc += v;
    cum[idx] = acc;
    xbf[idx] = (__bf16)v;
  }
  chunksum[((size_t)b * 16 + chnk) * DM + d] = acc;
}

// grid = 8 * 4 = 32 blocks; in-place exclusive scan over the 16 chunks
__global__ __launch_bounds__(256) void chunk_scan_kernel(float* __restrict__ chunksum) {
  int blk = blockIdx.x;
  int b   = blk >> 2;
  int d   = (blk & 3) * 256 + threadIdx.x;
  float run = 0.f;
  for (int c = 0; c < 16; ++c) {
    size_t i = ((size_t)b * 16 + c) * DM + d;
    float t = chunksum[i];
    chunksum[i] = run;
    run += t;
  }
}

// -------------------------------- stage 2: finalize avg + LayerNorm (bf16 h)
__global__ __launch_bounds__(256) void ln_finalize_kernel(float* __restrict__ cum,
                                                          const float* __restrict__ choff,
                                                          const float* __restrict__ g,
                                                          const float* __restrict__ bb,
                                                          __bf16* __restrict__ hbf) {
  __shared__ float2 red[256];
  int row  = blockIdx.x;
  int b    = row >> 11;
  int s    = row & (SEQ - 1);
  int chnk = s >> 7;
  int tid  = threadIdx.x;
  int d0   = tid * 4;
  size_t ri = (size_t)row * DM;

  float4 v = *(const float4*)(cum + ri + d0);
  float4 o = *(const float4*)(choff + ((size_t)b * 16 + chnk) * DM + d0);
  float inv = 1.0f / (float)(s + 1);
  v.x = (v.x + o.x) * inv;
  v.y = (v.y + o.y) * inv;
  v.z = (v.z + o.z) * inv;
  v.w = (v.w + o.w) * inv;

  float sum = v.x + v.y + v.z + v.w;
  float sq  = v.x * v.x + v.y * v.y + v.z * v.z + v.w * v.w;
  red[tid] = make_float2(sum, sq);
  __syncthreads();
  for (int off = 128; off > 0; off >>= 1) {
    if (tid < off) {
      red[tid].x += red[tid + off].x;
      red[tid].y += red[tid + off].y;
    }
    __syncthreads();
  }
  float mu  = red[0].x * (1.0f / (float)DM);
  float var = red[0].y * (1.0f / (float)DM) - mu * mu;
  float rs  = rsqrtf(var + 1e-6f);

  *(float4*)(cum + ri + d0) = v;  // final avg (fp32) in place

  float4 gg = *(const float4*)(g + d0);
  float4 bv = *(const float4*)(bb + d0);
  hbf[ri + d0 + 0] = (__bf16)((v.x - mu) * rs * gg.x + bv.x);
  hbf[ri + d0 + 1] = (__bf16)((v.y - mu) * rs * gg.y + bv.y);
  hbf[ri + d0 + 2] = (__bf16)((v.z - mu) * rs * gg.z + bv.z);
  hbf[ri + d0 + 3] = (__bf16)((v.w - mu) * rs * gg.w + bv.w);
}

// ----------------------------------------- stage 3/4: WMMA GEMM (FF layers)
// C[M,N] = A[M,K] * BT[N,K]^T; M=16384, N=K=1024; BM=128 BN=128 BK=32
// async double-buffered LDS staging; 8 waves as 2x4, each 64x32 = 4x2 wmma
// A-fragments loaded just-in-time (one live at a time) to relieve regalloc.
// MODE 0: out_bf = bf16(relu(acc+bias));  MODE 1: v=acc+bias+resid -> f32+bf16
template <int MODE>
__global__ __launch_bounds__(256) void gemm_ff_kernel(const __bf16* __restrict__ A,
                                                      const __bf16* __restrict__ BT,
                                                      const float* __restrict__ bias,
                                                      const float* __restrict__ resid,
                                                      __bf16* __restrict__ outBf,
                                                      float* __restrict__ outF) {
  const int K = DM, N = DM;
  __shared__ __bf16 lA[2][128 * 40];  // [row][k] k-contiguous, pad 40
  __shared__ __bf16 lB[2][128 * 40];  // [n][k]   k-contiguous, pad 40

  int tid = threadIdx.x;
  int rowBase = blockIdx.x * 128;
  int n0 = blockIdx.y * 128;
  int wid = tid >> 5, lane = tid & 31, half = lane >> 4, l15 = lane & 15;
  int waveM = wid >> 2, waveN = wid & 3;

  // per-thread staging chunks: A 2x16B, B 2x16B
  int ra0 = tid >> 2,          ka0 = (tid & 3) << 3;
  int ra1 = (tid + 256) >> 2,  ka1 = ((tid + 256) & 3) << 3;

  auto issue = [&](int k0, int buf) {
    async_cp16(A  + (size_t)(rowBase + ra0) * K + k0 + ka0, &lA[buf][ra0 * 40 + ka0]);
    async_cp16(A  + (size_t)(rowBase + ra1) * K + k0 + ka1, &lA[buf][ra1 * 40 + ka1]);
    async_cp16(BT + (size_t)(n0     + ra0) * K + k0 + ka0, &lB[buf][ra0 * 40 + ka0]);
    async_cp16(BT + (size_t)(n0     + ra1) * K + k0 + ka1, &lB[buf][ra1 * 40 + ka1]);
  };

  v8f zero = {0.f, 0.f, 0.f, 0.f, 0.f, 0.f, 0.f, 0.f};
  v8f acc[4][2];
#pragma unroll
  for (int mi = 0; mi < 4; ++mi)
#pragma unroll
    for (int ni = 0; ni < 2; ++ni) acc[mi][ni] = zero;

  const int NT = K / 32;
  issue(0, 0);
  async_wait0();
  __syncthreads();

  for (int t = 0; t < NT; ++t) {
    int buf = t & 1;
    if (t + 1 < NT) issue((t + 1) * 32, (t + 1) & 1);

    AFrag bfr[2];
#pragma unroll
    for (int ni = 0; ni < 2; ++ni) {
      const __bf16* bp = &lB[buf][(waveN * 32 + ni * 16 + l15) * 40 + half * 16];
      bfr[ni].q[0] = *(const uint4*)(bp);
      bfr[ni].q[1] = *(const uint4*)(bp + 8);
    }
#pragma unroll
    for (int mi = 0; mi < 4; ++mi) {
      AFrag afr;  // just-in-time: only one A fragment live at a time
      const __bf16* ap = &lA[buf][(waveM * 64 + mi * 16 + l15) * 40 + half * 16];
      afr.q[0] = *(const uint4*)(ap);
      afr.q[1] = *(const uint4*)(ap + 8);
#pragma unroll
      for (int ni = 0; ni < 2; ++ni)
        acc[mi][ni] = __builtin_amdgcn_wmma_f32_16x16x32_bf16(
            false, afr.v, false, bfr[ni].v, (short)0, acc[mi][ni], false, false);
    }

    async_wait0();
    __syncthreads();
  }

#pragma unroll
  for (int mi = 0; mi < 4; ++mi)
#pragma unroll
    for (int ni = 0; ni < 2; ++ni) {
      int gcol = n0 + waveN * 32 + ni * 16 + l15;
      float bv = bias[gcol];
#pragma unroll
      for (int i = 0; i < 8; ++i) {
        int grow = rowBase + waveM * 64 + mi * 16 + half * 8 + i;
        size_t oi = (size_t)grow * N + gcol;
        float v = acc[mi][ni][i] + bv;
        if (MODE == 0) {
          v = fmaxf(v, 0.f);
          outBf[oi] = (__bf16)v;
        } else {
          v += resid[oi];
          outF[oi] = v;
          outBf[oi] = (__bf16)v;
        }
      }
    }
}

// ------------------------- stage 5: fused gating GEMM + sigmoid-gate epilogue
// K=2048 over cat[x|avg_out]; wgT is bf16 [2048(N)][2048(K)]
// BM=128, BN=64 per gate set; 8 waves as 4x2, each 32x32 per set
__global__ __launch_bounds__(256) void gemm_gate_kernel(const __bf16* __restrict__ xbf,
                                                        const __bf16* __restrict__ abf,
                                                        const __bf16* __restrict__ wgT,
                                                        const float* __restrict__ bg,
                                                        const float* __restrict__ xf,
                                                        const float* __restrict__ aoutf,
                                                        float* __restrict__ gated) {
  const int KT = 2 * DM;
  __shared__ __bf16 lA[2][128 * 40];
  __shared__ __bf16 lBi[2][64 * 40];
  __shared__ __bf16 lBf[2][64 * 40];

  int tid = threadIdx.x;
  int rowBase = blockIdx.x * 128;
  int j0 = blockIdx.y * 64;
  int wid = tid >> 5, lane = tid & 31, half = lane >> 4, l15 = lane & 15;
  int waveM = wid >> 1, waveN = wid & 1;

  int ra0 = tid >> 2,         ka0 = (tid & 3) << 3;
  int ra1 = (tid + 256) >> 2, ka1 = ((tid + 256) & 3) << 3;

  auto issue = [&](int k0, int buf) {
    const __bf16* Asrc = (k0 < DM) ? (xbf + k0) : (abf + (k0 - DM));
    async_cp16(Asrc + (size_t)(rowBase + ra0) * DM + ka0, &lA[buf][ra0 * 40 + ka0]);
    async_cp16(Asrc + (size_t)(rowBase + ra1) * DM + ka1, &lA[buf][ra1 * 40 + ka1]);
    async_cp16(wgT + (size_t)(j0 + ra0) * KT + k0 + ka0,      &lBi[buf][ra0 * 40 + ka0]);
    async_cp16(wgT + (size_t)(j0 + DM + ra0) * KT + k0 + ka0, &lBf[buf][ra0 * 40 + ka0]);
  };

  v8f zero = {0.f, 0.f, 0.f, 0.f, 0.f, 0.f, 0.f, 0.f};
  v8f accI[2][2], accF[2][2];
#pragma unroll
  for (int mi = 0; mi < 2; ++mi)
#pragma unroll
    for (int ni = 0; ni < 2; ++ni) { accI[mi][ni] = zero; accF[mi][ni] = zero; }

  const int NT = KT / 32;
  issue(0, 0);
  async_wait0();
  __syncthreads();

  for (int t = 0; t < NT; ++t) {
    int buf = t & 1;
    if (t + 1 < NT) issue((t + 1) * 32, (t + 1) & 1);

    AFrag bi[2], bfg[2];
#pragma unroll
    for (int ni = 0; ni < 2; ++ni) {
      const __bf16* bpi = &lBi[buf][(waveN * 32 + ni * 16 + l15) * 40 + half * 16];
      bi[ni].q[0] = *(const uint4*)(bpi);
      bi[ni].q[1] = *(const uint4*)(bpi + 8);
      const __bf16* bpf = &lBf[buf][(waveN * 32 + ni * 16 + l15) * 40 + half * 16];
      bfg[ni].q[0] = *(const uint4*)(bpf);
      bfg[ni].q[1] = *(const uint4*)(bpf + 8);
    }
#pragma unroll
    for (int mi = 0; mi < 2; ++mi) {
      AFrag afr;  // just-in-time A fragment
      const __bf16* ap = &lA[buf][(waveM * 32 + mi * 16 + l15) * 40 + half * 16];
      afr.q[0] = *(const uint4*)(ap);
      afr.q[1] = *(const uint4*)(ap + 8);
#pragma unroll
      for (int ni = 0; ni < 2; ++ni) {
        accI[mi][ni] = __builtin_amdgcn_wmma_f32_16x16x32_bf16(
            false, afr.v, false, bi[ni].v, (short)0, accI[mi][ni], false, false);
        accF[mi][ni] = __builtin_amdgcn_wmma_f32_16x16x32_bf16(
            false, afr.v, false, bfg[ni].v, (short)0, accF[mi][ni], false, false);
      }
    }

    async_wait0();
    __syncthreads();
  }

#pragma unroll
  for (int mi = 0; mi < 2; ++mi)
#pragma unroll
    for (int ni = 0; ni < 2; ++ni) {
      int gcol = j0 + waveN * 32 + ni * 16 + l15;
      float bgi = bg[gcol];
      float bgf = bg[gcol + DM];
#pragma unroll
      for (int i = 0; i < 8; ++i) {
        int grow = rowBase + waveM * 32 + mi * 16 + half * 8 + i;
        size_t oi = (size_t)grow * DM + gcol;
        float gi = accI[mi][ni][i] + bgi;
        float gf = accF[mi][ni][i] + bgf;
        float si = 1.0f / (1.0f + expf(-gi));
        float sf = 1.0f / (1.0f + expf(-gf));
        gated[oi] = si * xf[oi] + sf * aoutf[oi];
      }
    }
}

// --------------------------------------------------------------------- host
extern "C" void kernel_launch(void* const* d_in, const int* in_sizes, int n_in,
                              void* d_out, int out_size, void* d_ws, size_t ws_size,
                              hipStream_t stream) {
  const float* x    = (const float*)d_in[0];
  const float* w1   = (const float*)d_in[1];
  const float* b1   = (const float*)d_in[2];
  const float* w2   = (const float*)d_in[3];
  const float* b2   = (const float*)d_in[4];
  const float* ln_g = (const float*)d_in[5];
  const float* ln_b = (const float*)d_in[6];
  const float* wg   = (const float*)d_in[7];
  const float* bg   = (const float*)d_in[8];

  float* gated = (float*)d_out;
  float* aoutF = gated + (size_t)NTOK * DM;

  char* p = (char*)d_ws;
  __bf16* xbf     = (__bf16*)p;  p += (size_t)NTOK * DM * 2;
  float*  avg     = (float*)p;   p += (size_t)NTOK * DM * 4;
  __bf16* hbf     = (__bf16*)p;  p += (size_t)NTOK * DM * 2;
  __bf16* interbf = (__bf16*)p;  p += (size_t)NTOK * DM * 2;
  __bf16* aoutbf  = (__bf16*)p;  p += (size_t)NTOK * DM * 2;
  __bf16* w1T     = (__bf16*)p;  p += (size_t)DM * DM * 2;
  __bf16* w2T     = (__bf16*)p;  p += (size_t)DM * DM * 2;
  __bf16* wgT     = (__bf16*)p;  p += (size_t)4 * DM * DM * 2;
  float*  chnk    = (float*)p;   p += (size_t)NBATCH * 16 * DM * 4;

  // 0) weights -> bf16, transposed to [N][K]
  transpose_bf16_kernel<<<dim3(DM / 32, DM / 32), 256, 0, stream>>>(w1, w1T, DM, DM);
  transpose_bf16_kernel<<<dim3(DM / 32, DM / 32), 256, 0, stream>>>(w2, w2T, DM, DM);
  transpose_bf16_kernel<<<dim3(2 * DM / 32, 2 * DM / 32), 256, 0, stream>>>(wg, wgT, 2 * DM, 2 * DM);

  // 1) chunked cumulative sum (+ x -> bf16)
  cumsum_chunk_kernel<<<512, 256, 0, stream>>>(x, avg, xbf, chnk);
  chunk_scan_kernel<<<32, 256, 0, stream>>>(chnk);

  // 2) finalize avg + LayerNorm -> h (bf16)
  ln_finalize_kernel<<<NTOK, 256, 0, stream>>>(avg, chnk, ln_g, ln_b, hbf);

  // 3) inter = relu(h @ w1 + b1)
  gemm_ff_kernel<0><<<dim3(NTOK / 128, DM / 128), 256, 0, stream>>>(
      hbf, w1T, b1, nullptr, interbf, nullptr);

  // 4) avg_out = inter @ w2 + b2 + avg
  gemm_ff_kernel<1><<<dim3(NTOK / 128, DM / 128), 256, 0, stream>>>(
      interbf, w2T, b2, avg, aoutbf, aoutF);

  // 5) gated = sigmoid(in_g)*x + sigmoid(fg_g)*avg_out
  gemm_gate_kernel<<<dim3(NTOK / 128, DM / 64), 256, 0, stream>>>(
      xbf, aoutbf, wgT, bg, x, aoutF, gated);
}
